// GraphConvLayer_42760694399467
// MI455X (gfx1250) — matrix operand order repaired and checked by reading, not compiled, not run on hardware
//
#include <hip/hip_runtime.h>
#include <hip/hip_bf16.h>

// ---------------------------------------------------------------------------
// GNN message-passing layer for MI455X (gfx1250, wave32, WMMA).
// fp32 storage -> bf16 WMMA operands (v_wmma_f32_16x16x32_bf16) -> fp32 acc,
// fp32 atomics for scatter. x pre-converted to bf16 once; on-the-fly packs
// use round-half-up + v_perm_b32. Two 16-row tiles per wave (each B fragment
// feeds 2 WMMAs), and B fragments are software-pipelined 2 deep so WMMAs
// never wait on a just-issued ds_load (avoids s_wait_dscnt 0x0 serialization
// seen in the previous round's codegen).
// ---------------------------------------------------------------------------

typedef __attribute__((ext_vector_type(16))) __bf16 v16bf;
typedef __attribute__((ext_vector_type(8)))  float  v8f;

union Frag16 {
    v16bf v;
    unsigned int u[8];
    uint4 q[2];
};

struct Bias8 { float v[8]; };

// pack two fp32 into one dword of 2 bf16 (low = first), round-half-up
__device__ __forceinline__ unsigned int pkbf2(float a, float b) {
    unsigned int ua = __float_as_uint(a) + 0x8000u;
    unsigned int ub = __float_as_uint(b) + 0x8000u;
#if __has_builtin(__builtin_amdgcn_perm)
    // D.b01 = a.b23 (low half), D.b23 = b.b23 (high half)
    return __builtin_amdgcn_perm(ub, ua, 0x07060302u);
#else
    return (ua >> 16) | (ub & 0xFFFF0000u);
#endif
}

__device__ __forceinline__ unsigned short f2bf(float f) {
    return (unsigned short)((__float_as_uint(f) + 0x8000u) >> 16);
}

// A-fragment (16x32 bf16, MxK) from row-major fp32 (converted on the fly).
// `p` points at row m, element (k_piece + kbase), kbase = 8*(lane>=16).
// Lane layout per ISA: v0..v3 = K kbase..kbase+7, v4..v7 = K kbase+16..kbase+23.
__device__ __forceinline__ v16bf ld_a_f32(const float* p) {
    float4 f0 = *(const float4*)(p);
    float4 f1 = *(const float4*)(p + 4);
    float4 f2 = *(const float4*)(p + 16);
    float4 f3 = *(const float4*)(p + 20);
    Frag16 f;
    f.u[0] = pkbf2(f0.x, f0.y); f.u[1] = pkbf2(f0.z, f0.w);
    f.u[2] = pkbf2(f1.x, f1.y); f.u[3] = pkbf2(f1.z, f1.w);
    f.u[4] = pkbf2(f2.x, f2.y); f.u[5] = pkbf2(f2.z, f2.w);
    f.u[6] = pkbf2(f3.x, f3.y); f.u[7] = pkbf2(f3.z, f3.w);
    return f.v;
}

// A-fragment from row-major bf16 (global or LDS): two 16B chunks 32B apart.
__device__ __forceinline__ v16bf ld_a_bf16(const unsigned short* p) {
    Frag16 f;
    f.q[0] = *(const uint4*)(p);
    f.q[1] = *(const uint4*)(p + 16);
    return f.v;
}

// B-fragment (32x16 bf16, KxN) from bf16 W transposed [N][K]:
// two 16B chunks 32B apart per lane (lanes 0-15: K k0..k0+7, k0+16..k0+23; lanes 16-31: +8).
__device__ __forceinline__ v16bf ld_b_bf16(const unsigned short* p) {
    Frag16 f;
    f.q[0] = *(const uint4*)(p);
    f.q[1] = *(const uint4*)(p + 16);
    return f.v;
}

#define WMMA_BF16(a, b, c) \
    __builtin_amdgcn_wmma_f32_16x16x32_bf16(false, (a), false, (b), (short)0, (c), false, false)

// One K-step across all 8 N-tiles with a 2-deep B-fragment pipeline:
// when WMMA(n) issues, B(n+1) and B(n+2) are already in flight.
template<int NT, int LDK>
__device__ __forceinline__ void gemm_nloop(const unsigned short* sW, int k0,
                                           int m, int hi,
                                           const v16bf (&a)[NT], v8f (&acc)[NT][8])
{
    const unsigned short* base = sW + m * LDK + k0 + hi * 8;
    v16bf b0 = ld_b_bf16(base);
    v16bf b1 = ld_b_bf16(base + 16 * LDK);
#pragma unroll
    for (int n = 0; n < 8; ++n) {
        v16bf bn;
        if (n < 6) bn = ld_b_bf16(base + (n + 2) * 16 * LDK);
#pragma unroll
        for (int u = 0; u < NT; ++u)
            acc[u][n] = WMMA_BF16(a[u], b0, acc[u][n]);
        b0 = b1;
        b1 = bn;
    }
}

// ---------------------------------------------------------------------------
// Prep kernels
// ---------------------------------------------------------------------------
__global__ void k_transpose_bf16(const float* __restrict__ src,
                                 unsigned short* __restrict__ dst,
                                 int K, int Nn) {
    int idx = blockIdx.x * blockDim.x + threadIdx.x;
    if (idx >= K * Nn) return;
    int k = idx / Nn;
    int n = idx - k * Nn;
    unsigned int u = __float_as_uint(src[idx]);
    u += 0x7FFFu + ((u >> 16) & 1u);            // RNE for weights (done once)
    dst[n * K + k] = (unsigned short)(u >> 16);
}

// cast fp32 -> bf16, 2 elements per thread (packed dword store)
__global__ void k_cast_bf16(const float* __restrict__ src,
                            unsigned int* __restrict__ dst, int npairs) {
    int idx = blockIdx.x * blockDim.x + threadIdx.x;
    if (idx >= npairs) return;
    float2 f = *(const float2*)(src + 2 * (size_t)idx);
    dst[idx] = pkbf2(f.x, f.y);
}

__global__ void k_zero_f32(float* __restrict__ p, int n) {
    int idx = blockIdx.x * blockDim.x + threadIdx.x;
    if (idx < n) p[idx] = 0.0f;
}

// ---------------------------------------------------------------------------
// Message path: NT 16-edge tiles per wave.
//   m_in[16x384] = [xbf[row] | xbf[col] | edge_attr]
//   h = relu(m_in @ W1m + b1m); msg = h @ W2m + b2m; agg[col] += msg
// ---------------------------------------------------------------------------
template<int NT>
__device__ __forceinline__ void msg_tiles(
    const unsigned short* __restrict__ xbf, const int* __restrict__ ei,
    const float* __restrict__ ea,
    const unsigned short* sW1, const unsigned short* sW2, unsigned short* sHw,
    Bias8 b1v, Bias8 b2v, float* __restrict__ agg,
    int E, int t0, int m, int hi, int kbase)
{
    const unsigned short* xR[NT];
    const unsigned short* xC[NT];
    const float* aE[NT];
    int cI[NT];
#pragma unroll
    for (int u = 0; u < NT; ++u) {
        const int e = (t0 + u) * 16 + m;
        const int rI = ei[e];
        cI[u] = ei[E + e];
        xR[u] = xbf + (size_t)rI    * 128;
        xC[u] = xbf + (size_t)cI[u] * 128;
        aE[u] = ea  + (size_t)e     * 128;
    }

    v8f acc[NT][8];
#pragma unroll
    for (int u = 0; u < NT; ++u)
#pragma unroll
        for (int n = 0; n < 8; ++n) acc[u][n] = v8f{};

    // GEMM1: K = 384 in 12 steps of 32. Pieces 0,1 are bf16 (pure loads),
    // piece 2 (edge_attr) converts fp32 on the fly.
#pragma unroll
    for (int s = 0; s < 12; ++s) {
        const int p  = (s & 3) << 5;
        const int k0 = s << 5;
        v16bf a[NT];
#pragma unroll
        for (int u = 0; u < NT; ++u) {
            if (s < 4)       a[u] = ld_a_bf16(xR[u] + p + kbase);
            else if (s < 8)  a[u] = ld_a_bf16(xC[u] + p + kbase);
            else             a[u] = ld_a_f32 (aE[u] + p + kbase);
        }
        gemm_nloop<NT, 384>(sW1, k0, m, hi, a, acc);
    }

    // bias + ReLU, restage h as bf16 row-major [16][128] per tile
#pragma unroll
    for (int u = 0; u < NT; ++u)
#pragma unroll
        for (int n = 0; n < 8; ++n) {
            const int nn = n * 16 + m;
#pragma unroll
            for (int r = 0; r < 8; ++r) {
                float h = acc[u][n][r] + b1v.v[n];
                h = h > 0.0f ? h : 0.0f;
                sHw[u * 2048 + (r + (hi << 3)) * 128 + nn] = f2bf(h);
            }
        }
    // wave-private LDS; DS ops are in-order within a wave -> no barrier

    // GEMM2: K = 128 in 4 steps
    v8f acc2[NT][8];
#pragma unroll
    for (int u = 0; u < NT; ++u)
#pragma unroll
        for (int n = 0; n < 8; ++n) acc2[u][n] = v8f{};
#pragma unroll
    for (int s = 0; s < 4; ++s) {
        const int k0 = s << 5;
        v16bf a[NT];
#pragma unroll
        for (int u = 0; u < NT; ++u)
            a[u] = ld_a_bf16(sHw + u * 2048 + m * 128 + k0 + kbase);
        gemm_nloop<NT, 128>(sW2, k0, m, hi, a, acc2);
    }

    // scatter (C/D layout: VGPR r -> row r+8*hi, col m)
#pragma unroll
    for (int u = 0; u < NT; ++u)
#pragma unroll
        for (int r = 0; r < 8; ++r) {
            const int node = __shfl(cI[u], r + (hi << 3));
            float* dst = agg + (size_t)node * 128;
#pragma unroll
            for (int n = 0; n < 8; ++n)
                atomicAdd(dst + n * 16 + m, acc2[u][n][r] + b2v.v[n]);
        }
}

__global__ void __launch_bounds__(256) k_msg(
    const unsigned short* __restrict__ xbf, const int* __restrict__ ei,
    const float* __restrict__ ea,
    const unsigned short* __restrict__ w1T, const unsigned short* __restrict__ w2T,
    const float* __restrict__ b1, const float* __restrict__ b2,
    float* __restrict__ agg, int E, int numTiles)
{
    extern __shared__ char smem[];
    unsigned short* sW1 = (unsigned short*)(smem);            // 384*128 bf16 [N][K]
    unsigned short* sW2 = (unsigned short*)(smem + 98304);    // 128*128 bf16 [N][K]
    unsigned short* sH  = (unsigned short*)(smem + 131072);   // 8 waves * 2 tiles * 4KB

    {
        const uint4* g1 = (const uint4*)w1T;  uint4* l1 = (uint4*)sW1;
        for (int i = threadIdx.x; i < 98304 / 16; i += blockDim.x) l1[i] = g1[i];
        const uint4* g2 = (const uint4*)w2T;  uint4* l2 = (uint4*)sW2;
        for (int i = threadIdx.x; i < 32768 / 16; i += blockDim.x) l2[i] = g2[i];
    }
    __syncthreads();

    const int lane  = threadIdx.x & 31;
    const int wave  = threadIdx.x >> 5;
    const int m     = lane & 15;
    const int hi    = lane >> 4;
    const int kbase = hi << 3;

    Bias8 b1v, b2v;
#pragma unroll
    for (int n = 0; n < 8; ++n) { b1v.v[n] = b1[n * 16 + m]; b2v.v[n] = b2[n * 16 + m]; }

    unsigned short* sHw = sH + wave * 4096;   // 2 tiles * 16*128 bf16

    const int numPairs = (numTiles + 1) >> 1;
    for (int pt = blockIdx.x * 8 + wave; pt < numPairs; pt += gridDim.x * 8) {
        const int t0 = pt * 2;
        if (t0 + 1 < numTiles)
            msg_tiles<2>(xbf, ei, ea, sW1, sW2, sHw, b1v, b2v, agg, E, t0, m, hi, kbase);
        else
            msg_tiles<1>(xbf, ei, ea, sW1, sW2, sHw, b1v, b2v, agg, E, t0, m, hi, kbase);
    }
}

// ---------------------------------------------------------------------------
// Update path: NT 16-node tiles per wave.
//   u_in[16x256] = [xbf | agg]; out = relu(u_in@W1u+b1u)@W2u + b2u + x
// ---------------------------------------------------------------------------
template<int NT>
__device__ __forceinline__ void upd_tiles(
    const unsigned short* __restrict__ xbf, const float* __restrict__ x,
    const float* __restrict__ agg,
    const unsigned short* sW1, const unsigned short* sW2, unsigned short* sHw,
    Bias8 b1v, Bias8 b2v, float* __restrict__ out,
    int t0, int m, int hi, int kbase)
{
    const unsigned short* aX[NT];
    const float* aG[NT];
#pragma unroll
    for (int u = 0; u < NT; ++u) {
        const int nd = (t0 + u) * 16 + m;
        aX[u] = xbf + (size_t)nd * 128;
        aG[u] = agg + (size_t)nd * 128;
    }

    v8f acc[NT][8];
#pragma unroll
    for (int u = 0; u < NT; ++u)
#pragma unroll
        for (int n = 0; n < 8; ++n) acc[u][n] = v8f{};

    // GEMM1: K = 256 in 8 steps (piece 0: bf16 x, piece 1: fp32 agg)
#pragma unroll
    for (int s = 0; s < 8; ++s) {
        const int p  = (s & 3) << 5;
        const int k0 = s << 5;
        v16bf a[NT];
#pragma unroll
        for (int u = 0; u < NT; ++u) {
            if (s < 4) a[u] = ld_a_bf16(aX[u] + p + kbase);
            else       a[u] = ld_a_f32 (aG[u] + p + kbase);
        }
        gemm_nloop<NT, 256>(sW1, k0, m, hi, a, acc);
    }

#pragma unroll
    for (int u = 0; u < NT; ++u)
#pragma unroll
        for (int n = 0; n < 8; ++n) {
            const int nn = n * 16 + m;
#pragma unroll
            for (int r = 0; r < 8; ++r) {
                float h = acc[u][n][r] + b1v.v[n];
                h = h > 0.0f ? h : 0.0f;
                sHw[u * 2048 + (r + (hi << 3)) * 128 + nn] = f2bf(h);
            }
        }

    v8f acc2[NT][8];
#pragma unroll
    for (int u = 0; u < NT; ++u)
#pragma unroll
        for (int n = 0; n < 8; ++n) acc2[u][n] = v8f{};
#pragma unroll
    for (int s = 0; s < 4; ++s) {
        const int k0 = s << 5;
        v16bf a[NT];
#pragma unroll
        for (int u = 0; u < NT; ++u)
            a[u] = ld_a_bf16(sHw + u * 2048 + m * 128 + k0 + kbase);
        gemm_nloop<NT, 128>(sW2, k0, m, hi, a, acc2);
    }

    // bias + residual (fp32 x) + store
#pragma unroll
    for (int u = 0; u < NT; ++u)
#pragma unroll
        for (int r = 0; r < 8; ++r) {
            const int row  = r + (hi << 3);
            const size_t o = (size_t)((t0 + u) * 16 + row) * 128;
#pragma unroll
            for (int n = 0; n < 8; ++n) {
                const int nn = n * 16 + m;
                out[o + nn] = acc2[u][n][r] + b2v.v[n] + x[o + nn];
            }
        }
}

__global__ void __launch_bounds__(256) k_upd(
    const unsigned short* __restrict__ xbf, const float* __restrict__ x,
    const float* __restrict__ agg,
    const unsigned short* __restrict__ w1T, const unsigned short* __restrict__ w2T,
    const float* __restrict__ b1, const float* __restrict__ b2,
    float* __restrict__ out, int numTiles)
{
    extern __shared__ char smem[];
    unsigned short* sW1 = (unsigned short*)(smem);            // 256*128 bf16 [N][K]
    unsigned short* sW2 = (unsigned short*)(smem + 65536);    // 128*128 bf16 [N][K]
    unsigned short* sH  = (unsigned short*)(smem + 98304);    // 8 waves * 2 tiles * 4KB

    {
        const uint4* g1 = (const uint4*)w1T;  uint4* l1 = (uint4*)sW1;
        for (int i = threadIdx.x; i < 65536 / 16; i += blockDim.x) l1[i] = g1[i];
        const uint4* g2 = (const uint4*)w2T;  uint4* l2 = (uint4*)sW2;
        for (int i = threadIdx.x; i < 32768 / 16; i += blockDim.x) l2[i] = g2[i];
    }
    __syncthreads();

    const int lane  = threadIdx.x & 31;
    const int wave  = threadIdx.x >> 5;
    const int m     = lane & 15;
    const int hi    = lane >> 4;
    const int kbase = hi << 3;

    Bias8 b1v, b2v;
#pragma unroll
    for (int n = 0; n < 8; ++n) { b1v.v[n] = b1[n * 16 + m]; b2v.v[n] = b2[n * 16 + m]; }

    unsigned short* sHw = sH + wave * 4096;

    const int numPairs = (numTiles + 1) >> 1;
    for (int pt = blockIdx.x * 8 + wave; pt < numPairs; pt += gridDim.x * 8) {
        const int t0 = pt * 2;
        if (t0 + 1 < numTiles)
            upd_tiles<2>(xbf, x, agg, sW1, sW2, sHw, b1v, b2v, out, t0, m, hi, kbase);
        else
            upd_tiles<1>(xbf, x, agg, sW1, sW2, sHw, b1v, b2v, out, t0, m, hi, kbase);
    }
}

// ---------------------------------------------------------------------------
// Host-side launch
// ---------------------------------------------------------------------------
extern "C" void kernel_launch(void* const* d_in, const int* in_sizes, int n_in,
                              void* d_out, int out_size, void* d_ws, size_t ws_size,
                              hipStream_t stream) {
    const float* x   = (const float*)d_in[0];
    const int*   ei  = (const int*)d_in[1];
    const float* ea  = (const float*)d_in[2];
    const float* W1m = (const float*)d_in[3];
    const float* b1m = (const float*)d_in[4];
    const float* W2m = (const float*)d_in[5];
    const float* b2m = (const float*)d_in[6];
    const float* W1u = (const float*)d_in[7];
    const float* b1u = (const float*)d_in[8];
    const float* W2u = (const float*)d_in[9];
    const float* b2u = (const float*)d_in[10];
    float* out = (float*)d_out;

    const int D  = 128;
    const int Nn = in_sizes[0] / D;       // 50000
    const int E  = in_sizes[1] / 2;       // 600000
    const int tilesE = E / 16;            // 37500
    const int tilesN = (Nn + 15) / 16;    // 3125

    // workspace layout (bytes, all 16B aligned)
    char* ws = (char*)d_ws;
    unsigned short* w1mT = (unsigned short*)(ws);             // 98304
    unsigned short* w2mT = (unsigned short*)(ws + 98304);     // 32768
    unsigned short* w1uT = (unsigned short*)(ws + 131072);    // 65536
    unsigned short* w2uT = (unsigned short*)(ws + 196608);    // 32768
    unsigned short* xbf  = (unsigned short*)(ws + 229376);    // N*128*2 = 12.8MB
    float*          agg  = (float*)(ws + 229376 + (size_t)Nn * D * 2);  // N*D*4

    // prep: bf16-transposed weights + bf16 copy of x
    k_transpose_bf16<<<(384 * 128 + 255) / 256, 256, 0, stream>>>(W1m, w1mT, 384, 128);
    k_transpose_bf16<<<(128 * 128 + 255) / 256, 256, 0, stream>>>(W2m, w2mT, 128, 128);
    k_transpose_bf16<<<(256 * 128 + 255) / 256, 256, 0, stream>>>(W1u, w1uT, 256, 128);
    k_transpose_bf16<<<(128 * 128 + 255) / 256, 256, 0, stream>>>(W2u, w2uT, 128, 128);
    {
        const int npairs = Nn * D / 2;
        k_cast_bf16<<<(npairs + 255) / 256, 256, 0, stream>>>(x, (unsigned int*)xbf, npairs);
    }

    // zero agg every call (atomic accumulation target)
    k_zero_f32<<<(Nn * D + 255) / 256, 256, 0, stream>>>(agg, Nn * D);

    // message + scatter  (LDS: 98304 + 32768 + 65536 = 196608)
    k_msg<<<1024, 256, 196608, stream>>>(
        xbf, ei, ea, w1mT, w2mT, b1m, b2m, agg, E, tilesE);

    // update + residual  (LDS: 65536 + 32768 + 65536 = 163840)
    k_upd<<<256, 256, 163840, stream>>>(
        xbf, x, agg, w1uT, w2uT, b1u, b2u, out, tilesN);
}
